// EncoderSRNN_82995948028593
// MI455X (gfx1250) — compile-verified, matrix operand order checked by reading, not run on hardware
//
#include <hip/hip_runtime.h>
#include <hip/hip_bf16.h>
#include <math.h>

// ---------------------------------------------------------------------------
// EncoderSRNN for MI455X (gfx1250, wave32, WMMA)
// S=256, B=512, H=256, NSTACK=2, SSIZE=64, ELEM=64
// ---------------------------------------------------------------------------

typedef __attribute__((ext_vector_type(16))) __bf16 v16bf;
typedef __attribute__((ext_vector_type(8)))  float  v8f;

#define WMMA_BF16(A,B,C) \
  __builtin_amdgcn_wmma_f32_16x16x32_bf16(false,(A),false,(B),(short)0,(C),false,false)

__device__ __forceinline__ v8f vzero8() {
  v8f z;
#pragma unroll
  for (int i = 0; i < 8; ++i) z[i] = 0.f;
  return z;
}

// Fragment loader for 16x16x32 bf16 WMMA.
// A-matrix:  base = tile (16 rows, row-major, stride in elems)
// B-matrix:  base = W + n0*stride (W row n = output column n; Y = X @ W^T)
// Layout per ISA 7.12.2: lanes 0-15 hold row/col (lane&15), K = {0..7,16..23};
// lanes 16-31 same row/col, K shifted by +8.
__device__ __forceinline__ v16bf load_frag16(const __bf16* base, int stride,
                                             int lane, int k0) {
  int row = lane & 15;
  int kb  = k0 + ((lane & 16) ? 8 : 0);
  const __bf16* p = base + row * stride + kb;
  union { uint4 q[2]; v16bf v; } u;
  u.q[0] = *(const uint4*)(p);
  u.q[1] = *(const uint4*)(p + 16);
  return u.v;
}

__device__ __forceinline__ float sigmoidf(float x) {
  return 1.f / (1.f + __expf(-x));
}

__device__ __forceinline__ float4 ld4(const float* p) { return *(const float4*)p; }
__device__ __forceinline__ void   st4(float* p, float4 v) { *(float4*)p = v; }
__device__ __forceinline__ float4 mix3(float pp, float4 a, float pq, float4 b,
                                       float pn, float4 c) {
  float4 r;
  r.x = pp * a.x + pq * b.x + pn * c.x;
  r.y = pp * a.y + pq * b.y + pn * c.y;
  r.z = pp * a.z + pq * b.z + pn * c.z;
  r.w = pp * a.w + pq * b.w + pn * c.w;
  return r;
}

// ---------------------------------------------------------------------------
// fp32 -> bf16 weight conversion
// ---------------------------------------------------------------------------
__global__ void cvt_bf16(const float* __restrict__ src, __bf16* __restrict__ dst, int n) {
  int i = blockIdx.x * 256 + threadIdx.x;
  if (i < n) dst[i] = (__bf16)src[i];
}

// ---------------------------------------------------------------------------
// stacks[b][n][s][e] = empty_elem[e]
// ---------------------------------------------------------------------------
__global__ void init_stacks(float* __restrict__ stacks, const float* __restrict__ empty_elem) {
  size_t i = (size_t)blockIdx.x * 256 + threadIdx.x;   // 4,194,304 total
  stacks[i] = empty_elem[i & 63];
}

// ---------------------------------------------------------------------------
// Persistent recurrence kernel: one block = 16 batch rows, all 256 steps.
// 256 threads = 8 waves; wave w owns H-columns [32w, 32w+32).
// ---------------------------------------------------------------------------
__global__ __launch_bounds__(256) void srnn_recur(
    const int*   __restrict__ tokens,     // [S,B]
    const float* __restrict__ emb,        // [VOCAB,H]
    const __bf16* __restrict__ Wih_bf,    // [768,256]
    const __bf16* __restrict__ Whh_bf,    // [768,256]
    const float* __restrict__ b_ih,       // [768]
    const float* __restrict__ b_hh,       // [768]
    const float* __restrict__ W_act,      // [6,256]
    const float* __restrict__ b_act,      // [6]
    const __bf16* __restrict__ Wstk_bf,   // [128,256]
    const float* __restrict__ b_stk,      // [128]
    const float* __restrict__ empty_elem, // [64]
    float* __restrict__ outputs,          // [S,B,H]
    float* __restrict__ stacks)           // [B,2,64,64] (in/out, in-place)
{
  __shared__ __bf16 x_bf[16][256];
  __shared__ __bf16 h_bf[16][256];
  __shared__ float  h_f [16][256];
  __shared__ float  p_lds[16][2][3];
  __shared__ float  push_lds[16][128];
  __shared__ float  bih_s[768], bhh_s[768], bstk_s[128], emp_s[64];
  __shared__ float  Wact_s[6 * 256], bact_s[6];

  const int tid  = threadIdx.x;
  const int lane = tid & 31;
  const int wave = tid >> 5;            // 0..7
  const int b0   = blockIdx.x * 16;     // batch tile base
  const int cw   = wave * 32;           // this wave's column base in H

  // cache constants, zero h
  for (int i = tid; i < 768; i += 256) { bih_s[i] = b_ih[i]; bhh_s[i] = b_hh[i]; }
  for (int i = tid; i < 128; i += 256) bstk_s[i] = b_stk[i];
  for (int i = tid; i < 1536; i += 256) Wact_s[i] = W_act[i];
  if (tid < 64) emp_s[tid] = empty_elem[tid];
  if (tid < 6)  bact_s[tid] = b_act[tid];
  for (int i = tid; i < 16 * 256; i += 256) {
    (&h_f[0][0])[i]  = 0.f;
    (&h_bf[0][0])[i] = (__bf16)0.f;
  }

  // stack-shift thread mapping: fixed (batch,stack,8-elem strip) per thread
  const int pr = tid >> 3;              // 0..31 -> (m, n)
  const int sm = pr & 15;
  const int sn = pr >> 4;
  const int se = (tid & 7) * 8;
  float* stk_base = stacks + (((size_t)(b0 + sm) * 2 + sn) * 64) * 64 + se;

  // per-wave weight base pointers (hoisted)
  const __bf16* Wi_base[6];
  const __bf16* Wh_base[6];
#pragma unroll
  for (int g = 0; g < 3; ++g)
#pragma unroll
    for (int tt = 0; tt < 2; ++tt) {
      int n = g * 256 + cw + tt * 16;
      Wi_base[g * 2 + tt] = Wih_bf + (size_t)n * 256;
      Wh_base[g * 2 + tt] = Whh_bf + (size_t)n * 256;
    }
  const __bf16* Ws_base = Wstk_bf + (size_t)(wave * 16) * 256;

  for (int t = 0; t < 256; ++t) {
    // ---- load & mask x_t into LDS (bf16) ----
    {
      int m = tid >> 4, c0 = (tid & 15) * 16;
      int tok = tokens[t * 512 + b0 + m];
      const float* er = emb + (size_t)tok * 256 + c0;
#pragma unroll
      for (int q = 0; q < 4; ++q) {
        float4 v = ld4(er + q * 4);
        if (tok == 0) { v.x = 0.f; v.y = 0.f; v.z = 0.f; v.w = 0.f; }
        x_bf[m][c0 + q * 4 + 0] = (__bf16)v.x;
        x_bf[m][c0 + q * 4 + 1] = (__bf16)v.y;
        x_bf[m][c0 + q * 4 + 2] = (__bf16)v.z;
        x_bf[m][c0 + q * 4 + 3] = (__bf16)v.w;
      }
    }
    __syncthreads();   // x ready; prev-step h/stack writes done

    // ---- stage A: all matmuls reading old h ----
    v8f acc_i[6], acc_h[6];              // [gate r/z/n][2 col-tiles]
#pragma unroll
    for (int i = 0; i < 6; ++i) { acc_i[i] = vzero8(); acc_h[i] = vzero8(); }
    v8f gp = vzero8();

    // NOTE: unroll 1 — keeps the live set at ~13 accums + 2 A-frags + 1 B-frag
    // so the allocator stays under 256 VGPRs with zero scratch spills.
#pragma unroll 1
    for (int k0 = 0; k0 < 256; k0 += 32) {
      v16bf ax = load_frag16(&x_bf[0][0], 256, lane, k0);
      v16bf ah = load_frag16(&h_bf[0][0], 256, lane, k0);
#pragma unroll
      for (int j = 0; j < 6; ++j) {
        v16bf wi = load_frag16(Wi_base[j], 256, lane, k0);
        acc_i[j] = WMMA_BF16(ax, wi, acc_i[j]);
        v16bf wh = load_frag16(Wh_base[j], 256, lane, k0);
        acc_h[j] = WMMA_BF16(ah, wh, acc_h[j]);
      }
      v16bf ws = load_frag16(Ws_base, 256, lane, k0);
      gp = WMMA_BF16(ah, ws, gp);
    }

    // push_vals = tanh(h @ Wstk^T + b_stk) ; wave w owns cols [16w,16w+16)
    {
      int Mb  = (lane & 16) ? 8 : 0;
      int col = wave * 16 + (lane & 15);
#pragma unroll
      for (int r = 0; r < 8; ++r)
        push_lds[r + Mb][col] = tanhf(gp[r] + bstk_s[col]);
    }

    // action softmax (uses OLD h), one wave: lane -> (row, stack)
    if (wave == 0) {
      int m = lane & 15, n = lane >> 4;
      float s0 = bact_s[n * 3 + 0], s1 = bact_s[n * 3 + 1], s2 = bact_s[n * 3 + 2];
#pragma unroll 4
      for (int k = 0; k < 256; ++k) {
        float hv = h_f[m][k];
        s0 += hv * Wact_s[(n * 3 + 0) * 256 + k];
        s1 += hv * Wact_s[(n * 3 + 1) * 256 + k];
        s2 += hv * Wact_s[(n * 3 + 2) * 256 + k];
      }
      float mx = fmaxf(s0, fmaxf(s1, s2));
      float e0 = __expf(s0 - mx), e1 = __expf(s1 - mx), e2 = __expf(s2 - mx);
      float inv = 1.f / (e0 + e1 + e2);
      p_lds[m][n][0] = e0 * inv;  // push
      p_lds[m][n][1] = e1 * inv;  // pop
      p_lds[m][n][2] = e2 * inv;  // noop
    }
    __syncthreads();   // h reads done, p/push published

    // ---- stage B: GRU gates, update h (each lane owns its own elements) ----
    {
      int Mb = (lane & 16) ? 8 : 0;
#pragma unroll
      for (int tt = 0; tt < 2; ++tt) {
        int colb = cw + tt * 16 + (lane & 15);
        float bir = bih_s[colb],        bhr = bhh_s[colb];
        float biz = bih_s[256 + colb],  bhz = bhh_s[256 + colb];
        float bin = bih_s[512 + colb],  bhn = bhh_s[512 + colb];
#pragma unroll
        for (int r = 0; r < 8; ++r) {
          int M = r + Mb;
          float ir  = acc_i[0 * 2 + tt][r] + bir;
          float hr  = acc_h[0 * 2 + tt][r] + bhr;
          float iz  = acc_i[1 * 2 + tt][r] + biz;
          float hz  = acc_h[1 * 2 + tt][r] + bhz;
          float in_ = acc_i[2 * 2 + tt][r] + bin;
          float hn  = acc_h[2 * 2 + tt][r] + bhn;
          float rg = sigmoidf(ir + hr);
          float zg = sigmoidf(iz + hz);
          float ng = tanhf(in_ + rg * hn);
          float hold = h_f[M][colb];
          float hnew = (1.f - zg) * ng + zg * hold;
          outputs[((size_t)t * 512 + b0 + M) * 256 + colb] = hnew;
          h_f[M][colb]  = hnew;
          h_bf[M][colb] = (__bf16)hnew;
        }
      }
    }

    // ---- stage C: in-place stack shift (rolling window, thread-private strip)
    {
      float pp = p_lds[sm][sn][0], pq = p_lds[sm][sn][1], pn = p_lds[sm][sn][2];
      float4 cur0 = ld4(stk_base),      cur1 = ld4(stk_base + 4);
      float4 nxt0 = ld4(stk_base + 64), nxt1 = ld4(stk_base + 68);
      // row 0 = p_push * push_vals
      {
        float4 w0, w1;
        const float* pv = &push_lds[sm][sn * 64 + se];
        w0.x = pp * pv[0]; w0.y = pp * pv[1]; w0.z = pp * pv[2]; w0.w = pp * pv[3];
        w1.x = pp * pv[4]; w1.y = pp * pv[5]; w1.z = pp * pv[6]; w1.w = pp * pv[7];
        st4(stk_base, w0); st4(stk_base + 4, w1);
      }
#pragma unroll 1
      for (int s = 1; s < 63; ++s) {
        float4 prv0 = cur0, prv1 = cur1;
        cur0 = nxt0; cur1 = nxt1;
        nxt0 = ld4(stk_base + (s + 1) * 64);
        nxt1 = ld4(stk_base + (s + 1) * 64 + 4);
        st4(stk_base + s * 64,     mix3(pp, prv0, pq, nxt0, pn, cur0));
        st4(stk_base + s * 64 + 4, mix3(pp, prv1, pq, nxt1, pn, cur1));
      }
      // row 63 = empty_elem
      float4 e0 = ld4(&emp_s[se]), e1 = ld4(&emp_s[se + 4]);
      st4(stk_base + 63 * 64, e0); st4(stk_base + 63 * 64 + 4, e1);
    }
    __syncthreads();
  }
}

// ---------------------------------------------------------------------------
// a_raw[row] = tanh(out[row] @ Wa1^T + ba1) . Wa2 + ba2, masked ; row = s*B+b
// One block = 16 rows; 8 waves x 2 N-tiles = 256 cols.
// ---------------------------------------------------------------------------
__global__ __launch_bounds__(256) void attn_score(
    const float* __restrict__ outputs, const __bf16* __restrict__ Wa1_bf,
    const float* __restrict__ ba1, const float* __restrict__ Wa2,
    const float* __restrict__ ba2, const int* __restrict__ tokens,
    float* __restrict__ a_raw)
{
  __shared__ __bf16 x_bf[16][256];
  __shared__ float  red[16];
  const int tid = threadIdx.x, lane = tid & 31, wave = tid >> 5;
  const size_t r0 = (size_t)blockIdx.x * 16;

  {
    int m = tid >> 4, c0 = (tid & 15) * 16;
    const float* src = outputs + (r0 + m) * 256 + c0;
#pragma unroll
    for (int q = 0; q < 4; ++q) {
      float4 v = ld4(src + q * 4);
      x_bf[m][c0 + q * 4 + 0] = (__bf16)v.x;
      x_bf[m][c0 + q * 4 + 1] = (__bf16)v.y;
      x_bf[m][c0 + q * 4 + 2] = (__bf16)v.z;
      x_bf[m][c0 + q * 4 + 3] = (__bf16)v.w;
    }
  }
  if (tid < 16) red[tid] = 0.f;
  __syncthreads();

  v8f acc0 = vzero8(), acc1 = vzero8();
  const int n0 = wave * 32;
  const __bf16* W0 = Wa1_bf + (size_t)n0 * 256;
  const __bf16* W1 = Wa1_bf + (size_t)(n0 + 16) * 256;
#pragma unroll 1
  for (int k0 = 0; k0 < 256; k0 += 32) {
    v16bf a  = load_frag16(&x_bf[0][0], 256, lane, k0);
    v16bf w0 = load_frag16(W0, 256, lane, k0);
    acc0 = WMMA_BF16(a, w0, acc0);
    v16bf w1 = load_frag16(W1, 256, lane, k0);
    acc1 = WMMA_BF16(a, w1, acc1);
  }

  int Mb = (lane & 16) ? 8 : 0;
#pragma unroll
  for (int tt = 0; tt < 2; ++tt) {
    v8f acc = tt ? acc1 : acc0;
    int col = n0 + tt * 16 + (lane & 15);
    float w2 = Wa2[col], bb = ba1[col];
#pragma unroll
    for (int r = 0; r < 8; ++r)
      atomicAdd(&red[r + Mb], tanhf(acc[r] + bb) * w2);   // ds_add_f32
  }
  __syncthreads();
  if (tid < 16) {
    size_t row = r0 + tid;
    int tok = tokens[row];                 // tokens[s,b] flattens the same way
    a_raw[row] = (tok == 0) ? -__builtin_inff() : (red[tid] + ba2[0]);
  }
}

// ---------------------------------------------------------------------------
// Per-batch softmax over S and weighted sum of outputs.
// ---------------------------------------------------------------------------
__global__ __launch_bounds__(256) void attn_pool(
    const float* __restrict__ outputs, const float* __restrict__ a_raw,
    float* __restrict__ final_hidden)
{
  const int b = blockIdx.x, tid = threadIdx.x;
  __shared__ float av[256], red[256];
  float x = a_raw[(size_t)tid * 512 + b];
  av[tid] = x; red[tid] = x;
  __syncthreads();
  for (int off = 128; off > 0; off >>= 1) {
    if (tid < off) red[tid] = fmaxf(red[tid], red[tid + off]);
    __syncthreads();
  }
  float mx = red[0];
  __syncthreads();
  float e = __expf(av[tid] - mx);
  red[tid] = e;
  __syncthreads();
  av[tid] = e;
  for (int off = 128; off > 0; off >>= 1) {
    if (tid < off) red[tid] += red[tid + off];
    __syncthreads();
  }
  float inv = 1.f / red[0];
  __syncthreads();
  float acc = 0.f;
#pragma unroll 4
  for (int s = 0; s < 256; ++s)
    acc += av[s] * outputs[((size_t)s * 512 + b) * 256 + tid];
  final_hidden[b * 256 + tid] = acc * inv;
}

// ---------------------------------------------------------------------------
extern "C" void kernel_launch(void* const* d_in, const int* in_sizes, int n_in,
                              void* d_out, int out_size, void* d_ws, size_t ws_size,
                              hipStream_t stream) {
  const int*   tokens     = (const int*)  d_in[0];
  const float* emb        = (const float*)d_in[1];
  const float* W_ih       = (const float*)d_in[2];
  const float* W_hh       = (const float*)d_in[3];
  const float* b_ih       = (const float*)d_in[4];
  const float* b_hh       = (const float*)d_in[5];
  const float* W_act      = (const float*)d_in[6];
  const float* b_act      = (const float*)d_in[7];
  const float* W_stk      = (const float*)d_in[8];
  const float* b_stk      = (const float*)d_in[9];
  const float* empty_elem = (const float*)d_in[10];
  // d_in[11], d_in[12]: W_up / W_down — pure shifts, implemented directly
  const float* Wa1        = (const float*)d_in[13];
  const float* ba1        = (const float*)d_in[14];
  const float* Wa2        = (const float*)d_in[15];
  const float* ba2        = (const float*)d_in[16];

  float* out          = (float*)d_out;
  float* outputs      = out;                                   // S*B*H
  float* final_hidden = out + (size_t)256 * 512 * 256;         // B*H
  float* stacks       = final_hidden + (size_t)512 * 256;      // B*2*64*64

  // workspace layout (all 16B-aligned)
  __bf16* Wih_bf  = (__bf16*)d_ws;               // 196608 elems
  __bf16* Whh_bf  = Wih_bf + 196608;             // 196608
  __bf16* Wstk_bf = Whh_bf + 196608;             // 32768
  __bf16* Wa1_bf  = Wstk_bf + 32768;             // 65536
  float*  a_raw   = (float*)(Wa1_bf + 65536);    // 131072 floats

  cvt_bf16<<<(196608 + 255) / 256, 256, 0, stream>>>(W_ih, Wih_bf, 196608);
  cvt_bf16<<<(196608 + 255) / 256, 256, 0, stream>>>(W_hh, Whh_bf, 196608);
  cvt_bf16<<<(32768  + 255) / 256, 256, 0, stream>>>(W_stk, Wstk_bf, 32768);
  cvt_bf16<<<(65536  + 255) / 256, 256, 0, stream>>>(Wa1,  Wa1_bf,  65536);

  init_stacks<<<(512 * 2 * 64 * 64) / 256, 256, 0, stream>>>(stacks, empty_elem);

  srnn_recur<<<512 / 16, 256, 0, stream>>>(
      tokens, emb, Wih_bf, Whh_bf, b_ih, b_hh, W_act, b_act,
      Wstk_bf, b_stk, empty_elem, outputs, stacks);

  attn_score<<<(256 * 512) / 16, 256, 0, stream>>>(
      outputs, Wa1_bf, ba1, Wa2, ba2, tokens, a_raw);

  attn_pool<<<512, 256, 0, stream>>>(outputs, a_raw, final_hidden);
}
